// MultiHeadAttention_72155450573080
// MI455X (gfx1250) — matrix operand order repaired and checked by reading, not compiled, not run on hardware
//
#include <hip/hip_runtime.h>

// ---------------------------------------------------------------------------
// MHA forward for B=2, S=2048, D=768, H=12, DK=64 on gfx1250 (CDNA5).
// All GEMMs via v_wmma_f32_16x16x32_bf16 (fp32 -> bf16 on the fly, f32 acc).
// Flash-style causal attention: logits never touch HBM. K tiles are streamed
// with double-buffered GLOBAL_LOAD_ASYNC_TO_LDS_B128 (ASYNCcnt).
// ---------------------------------------------------------------------------

typedef unsigned short u16;
typedef unsigned int   u32;
typedef unsigned long long u64;

typedef __bf16 bf16x16 __attribute__((ext_vector_type(16)));
typedef float  f32x8   __attribute__((ext_vector_type(8)));
typedef float  f32x4   __attribute__((ext_vector_type(4)));
typedef u32    u32x4   __attribute__((ext_vector_type(4)));
typedef u32    u32x2   __attribute__((ext_vector_type(2)));

union Frag { bf16x16 v; u32x4 u[2]; };

static constexpr int kB = 2, kS = 2048, kD = 768, kH = 12, kDK = 64;
static constexpr float kScale = 0.036084391824351615f;  // 1/sqrt(768)

__device__ __forceinline__ u32 f2bf(float f) {
  u32 u = __float_as_uint(f);
  return (u + 0x7FFFu + ((u >> 16) & 1u)) >> 16;  // round-to-nearest-even
}

__device__ __forceinline__ f32x8 wmma_bf16(const Frag& a, const Frag& b, f32x8 c) {
  return __builtin_amdgcn_wmma_f32_16x16x32_bf16(
      false, a.v, false, b.v, (short)0, c, false, false);
}

// Async global->LDS copy of 16 bytes (CDNA5, ASYNCcnt-tracked, no VGPR data).
__device__ __forceinline__ void async_copy_b128(u32 lds_byte_addr, u64 gaddr) {
  asm volatile("global_load_async_to_lds_b128 %0, %1, off"
               :: "v"(lds_byte_addr), "v"(gaddr) : "memory");
}
__device__ __forceinline__ void wait_asynccnt0() {
  asm volatile("s_wait_asynccnt 0x0" ::: "memory");
}

// ---------------------------------------------------------------------------
// QKV projection: X[4096,768] @ W[768,768] + bias -> bf16 [B,H,S,DK]
// Block tile 128x128, 8 waves (2x4), K-step 32.
// ---------------------------------------------------------------------------
__global__ __launch_bounds__(256) void proj_qkv_kernel(
    const float* __restrict__ X, const float* __restrict__ W,
    const float* __restrict__ bias, u16* __restrict__ dst) {
  __shared__ u32x4 lA[512];  // [m:128][k:32] bf16, row-major
  __shared__ u32x4 lB[512];  // [n:128][k:32] bf16 (transposed W)

  const int tid  = threadIdx.x;
  const int lane = tid & 31, wave = tid >> 5;
  const int l15  = lane & 15, hi = lane >> 4;
  const int wm   = wave >> 2, wn = wave & 3;
  const int m0   = blockIdx.y * 128, n0 = blockIdx.x * 128;

  const f32x8 zf = {0.f, 0.f, 0.f, 0.f, 0.f, 0.f, 0.f, 0.f};
  f32x8 acc[4][2];
#pragma unroll
  for (int i = 0; i < 4; ++i)
#pragma unroll
    for (int j = 0; j < 2; ++j) acc[i][j] = zf;

  for (int k0 = 0; k0 < kD; k0 += 32) {
    __syncthreads();
    // Stage A tile: fp32 -> bf16
#pragma unroll
    for (int i = 0; i < 4; ++i) {
      int idx = tid + i * 256;
      int row = idx >> 3, c4 = idx & 7;
      const float* ap = X + (size_t)(m0 + row) * kD + k0 + c4 * 4;
      f32x4 vv = *(const f32x4*)ap;
      if (k0 + 32 < kD) __builtin_prefetch(ap + 32, 0, 1);
      u32x2 p;
      p[0] = f2bf(vv[0]) | (f2bf(vv[1]) << 16);
      p[1] = f2bf(vv[2]) | (f2bf(vv[3]) << 16);
      ((u32x2*)lA)[row * 8 + c4] = p;
    }
    // Stage B tile transposed: W[k][n] -> lB[n][k], fp32 -> bf16
    u16* lBs = (u16*)lB;
#pragma unroll
    for (int i = 0; i < 4; ++i) {
      int idx = tid + i * 256;
      int kk = idx >> 5, c4 = idx & 31;
      const float* wp = W + (size_t)(k0 + kk) * kD + n0 + c4 * 4;
      f32x4 vv = *(const f32x4*)wp;
      if (k0 + 32 < kD) __builtin_prefetch(wp + 32 * kD, 0, 1);
#pragma unroll
      for (int j = 0; j < 4; ++j) lBs[(c4 * 4 + j) * 32 + kk] = (u16)f2bf(vv[j]);
    }
    __syncthreads();

    Frag bfr[2];
#pragma unroll
    for (int ci = 0; ci < 2; ++ci) {
      int col = wn * 32 + ci * 16 + l15;
      bfr[ci].u[0] = lB[col * 4 + 2 * hi];
      bfr[ci].u[1] = lB[col * 4 + 2 * hi + 1];
    }
#pragma unroll
    for (int ri = 0; ri < 4; ++ri) {
      Frag a;
      int row = wm * 64 + ri * 16 + l15;
      a.u[0] = lA[row * 4 + 2 * hi];
      a.u[1] = lA[row * 4 + 2 * hi + 1];
#pragma unroll
      for (int ci = 0; ci < 2; ++ci) acc[ri][ci] = wmma_bf16(a, bfr[ci], acc[ri][ci]);
    }
  }

  // Epilogue: +bias, head-split bf16 store [B,H,S,DK]
#pragma unroll
  for (int ri = 0; ri < 4; ++ri) {
#pragma unroll
    for (int ci = 0; ci < 2; ++ci) {
      int gn = n0 + wn * 32 + ci * 16 + l15;
      int h = gn >> 6, dk = gn & 63;
      float bb = bias[gn];
#pragma unroll
      for (int r = 0; r < 8; ++r) {
        int gm = m0 + wm * 64 + ri * 16 + r + 8 * hi;
        int bi = gm >> 11, s = gm & (kS - 1);
        float val = acc[ri][ci][r] + bb;
        dst[((size_t)((bi * kH + h) * kS + s) << 6) + dk] = (u16)f2bf(val);
      }
    }
  }
}

// ---------------------------------------------------------------------------
// Flash attention, causal. Block = 128 q rows (8 waves x 16 rows), streams
// 32-key tiles. K tiles: double-buffered async global->LDS DMA (row-major
// layout doubles as the QK^T B-operand). V staged transposed via VGPRs.
// ---------------------------------------------------------------------------
__global__ __launch_bounds__(256) void attn_kernel(
    const u16* __restrict__ Qh, const u16* __restrict__ Kh,
    const u16* __restrict__ Vh, u16* __restrict__ attnW) {
  __shared__ u32x4 lK2[2][256];  // ping-pong [key:32][dk:64] bf16
  __shared__ u32x4 lV[256];      // [dk:64][key:32] bf16 (4 u32x4 per dk row)
  __shared__ u32x4 lP[8 * 64];   // per-wave P tile [q:16][key:32] bf16

  const int tid  = threadIdx.x;
  const int lane = tid & 31, wave = tid >> 5;
  const int l15  = lane & 15, hi = lane >> 4;
  const int bh   = blockIdx.y;           // 0..23
  const int bq   = blockIdx.x;           // 0..15 (128-row q blocks)
  const int q0w  = bq * 128 + wave * 16; // this wave's 16 q rows
  const int key  = tid >> 3, cc = tid & 7;

  // Q fragments straight from global (bf16, row pitch 64 = 8 u32x4)
  const u32x4* Q4 = (const u32x4*)Qh;
  const int qrow = q0w + l15;
  const int base8 = (bh * kS + qrow) * 8;
  Frag aq[2];
#pragma unroll
  for (int c = 0; c < 2; ++c) {
    aq[c].u[0] = Q4[base8 + 4 * c + 2 * hi];
    aq[c].u[1] = Q4[base8 + 4 * c + 2 * hi + 1];
  }

  const f32x8 zf = {0.f, 0.f, 0.f, 0.f, 0.f, 0.f, 0.f, 0.f};
  f32x8 accv[4];
  float mrow[8], lrow[8];
#pragma unroll
  for (int t = 0; t < 4; ++t) accv[t] = zf;
#pragma unroll
  for (int r = 0; r < 8; ++r) { mrow[r] = -1e30f; lrow[r] = 0.f; }

  const int nkt = (bq + 1) * 4;  // causal: keys up to q0_block+127

  // Prologue: async-DMA K tile 0 into buffer 0 (one b128 per thread).
  {
    u64 ga = (u64)(const void*)(((const u32x4*)Kh) + (bh * kS + key) * 8 + cc);
    u32 la = (u32)(size_t)(const void*)&lK2[0][tid];
    async_copy_b128(la, ga);
  }

  for (int kt = 0; kt < nkt; ++kt) {
    const int kb0 = kt * 32;
    const int buf = kt & 1;
    __syncthreads();  // previous tile's LDS readers done (frees lV, lK2[buf^1], lP)
    {
      // V tile: load + transpose to [dk][key]; prefetch next tile.
      int gidx = (bh * kS + kb0 + key) * 8 + cc;
      u32x4 vv = ((const u32x4*)Vh)[gidx];
      if (kt + 1 < nkt) __builtin_prefetch((const void*)(((const u32x4*)Vh) + gidx + 256), 0, 1);
      u16* lVs = (u16*)lV;
#pragma unroll
      for (int j = 0; j < 4; ++j) {
        u32 w = vv[j];
        lVs[(cc * 8 + 2 * j) * 32 + key]     = (u16)(w & 0xffffu);
        lVs[(cc * 8 + 2 * j + 1) * 32 + key] = (u16)(w >> 16);
      }
    }
    // K(kt) has landed; immediately launch the DMA for K(kt+1).
    wait_asynccnt0();
    if (kt + 1 < nkt) {
      u64 ga = (u64)(const void*)(((const u32x4*)Kh) + (bh * kS + kb0 + 32 + key) * 8 + cc);
      u32 la = (u32)(size_t)(const void*)&lK2[buf ^ 1][tid];
      async_copy_b128(la, ga);
    }
    __syncthreads();  // publish V transpose + K arrival to all waves
    const u32x4* lK = lK2[buf];

    // Scores: S = Q K^T  (two 16-key column tiles, two 32-dk k-steps)
    f32x8 s0 = zf, s1 = zf;
#pragma unroll
    for (int c = 0; c < 2; ++c) {
      Frag b0, b1;
      int n0r = l15, n1r = 16 + l15;
      b0.u[0] = lK[n0r * 8 + 4 * c + 2 * hi];
      b0.u[1] = lK[n0r * 8 + 4 * c + 2 * hi + 1];
      b1.u[0] = lK[n1r * 8 + 4 * c + 2 * hi];
      b1.u[1] = lK[n1r * 8 + 4 * c + 2 * hi + 1];
      s0 = wmma_bf16(aq[c], b0, s0);
      s1 = wmma_bf16(aq[c], b1, s1);
    }

    // Online softmax (rows stripe across 16-lane halves -> shfl_xor width 16)
    float fac[8];
    const int kg0 = kb0 + l15, kg1 = kg0 + 16;
    u16* lPs = (u16*)(lP + wave * 64);
#pragma unroll
    for (int r = 0; r < 8; ++r) {
      int qi = q0w + r + 8 * hi;
      float x0 = s0[r] * kScale + (kg0 > qi ? -1e9f : 0.f);
      float x1 = s1[r] * kScale + (kg1 > qi ? -1e9f : 0.f);
      float tm = fmaxf(x0, x1);
      tm = fmaxf(tm, __shfl_xor(tm, 1, 16));
      tm = fmaxf(tm, __shfl_xor(tm, 2, 16));
      tm = fmaxf(tm, __shfl_xor(tm, 4, 16));
      tm = fmaxf(tm, __shfl_xor(tm, 8, 16));
      float mn = fmaxf(mrow[r], tm);
      float f  = __expf(mrow[r] - mn);
      float p0 = __expf(x0 - mn);
      float p1 = __expf(x1 - mn);
      float rs = p0 + p1;
      rs += __shfl_xor(rs, 1, 16);
      rs += __shfl_xor(rs, 2, 16);
      rs += __shfl_xor(rs, 4, 16);
      rs += __shfl_xor(rs, 8, 16);
      lrow[r] = lrow[r] * f + rs;
      mrow[r] = mn;
      fac[r] = f;
      int m = r + 8 * hi;  // C-layout -> A-layout via LDS
      lPs[m * 32 + l15]      = (u16)f2bf(p0);
      lPs[m * 32 + 16 + l15] = (u16)f2bf(p1);
    }
    // Rescale running output
#pragma unroll
    for (int t = 0; t < 4; ++t)
#pragma unroll
      for (int r = 0; r < 8; ++r) accv[t][r] *= fac[r];

    __syncthreads();  // publish P (per-wave region, cross-lane) before reads

    // O += P V  (k-dim = 32 keys, one WMMA k-step; 4 dk column tiles)
    Frag ap;
    const u32x4* lPw = lP + wave * 64;
    ap.u[0] = lPw[l15 * 4 + 2 * hi];
    ap.u[1] = lPw[l15 * 4 + 2 * hi + 1];
#pragma unroll
    for (int t = 0; t < 4; ++t) {
      Frag bv;
      int n = t * 16 + l15;
      bv.u[0] = lV[n * 4 + 2 * hi];
      bv.u[1] = lV[n * 4 + 2 * hi + 1];
      accv[t] = wmma_bf16(ap, bv, accv[t]);
    }
  }

  // Epilogue: normalize, merge heads -> bf16 [B,S,D]
  const int bi = bh / kH, h = bh % kH;
#pragma unroll
  for (int t = 0; t < 4; ++t) {
    int dk = h * 64 + t * 16 + l15;
#pragma unroll
    for (int r = 0; r < 8; ++r) {
      int qi = q0w + r + 8 * hi;
      float val = accv[t][r] / lrow[r];
      attnW[(size_t)(bi * kS + qi) * kD + dk] = (u16)f2bf(val);
    }
  }
}

// ---------------------------------------------------------------------------
// Output projection: attn(bf16)[4096,768] @ Wo + bo, ReLU, fp32 out
// ---------------------------------------------------------------------------
__global__ __launch_bounds__(256) void out_proj_kernel(
    const u16* __restrict__ Abf, const float* __restrict__ W,
    const float* __restrict__ bias, float* __restrict__ out) {
  __shared__ u32x4 lA[512];
  __shared__ u32x4 lB[512];

  const int tid  = threadIdx.x;
  const int lane = tid & 31, wave = tid >> 5;
  const int l15  = lane & 15, hi = lane >> 4;
  const int wm   = wave >> 2, wn = wave & 3;
  const int m0   = blockIdx.y * 128, n0 = blockIdx.x * 128;

  const f32x8 zf = {0.f, 0.f, 0.f, 0.f, 0.f, 0.f, 0.f, 0.f};
  f32x8 acc[4][2];
#pragma unroll
  for (int i = 0; i < 4; ++i)
#pragma unroll
    for (int j = 0; j < 2; ++j) acc[i][j] = zf;

  const u32x4* A4 = (const u32x4*)Abf;  // 96 u32x4 per row
  for (int k0 = 0; k0 < kD; k0 += 32) {
    __syncthreads();
#pragma unroll
    for (int i = 0; i < 2; ++i) {
      int idx = tid + i * 256;
      int row = idx >> 2, c = idx & 3;
      const u32x4* ap = A4 + (size_t)(m0 + row) * 96 + (k0 >> 3) + c;
      lA[row * 4 + c] = *ap;
      if (k0 + 32 < kD) __builtin_prefetch((const void*)(ap + 4), 0, 1);
    }
    u16* lBs = (u16*)lB;
#pragma unroll
    for (int i = 0; i < 4; ++i) {
      int idx = tid + i * 256;
      int kk = idx >> 5, c4 = idx & 31;
      const float* wp = W + (size_t)(k0 + kk) * kD + n0 + c4 * 4;
      f32x4 vv = *(const f32x4*)wp;
      if (k0 + 32 < kD) __builtin_prefetch(wp + 32 * kD, 0, 1);
#pragma unroll
      for (int j = 0; j < 4; ++j) lBs[(c4 * 4 + j) * 32 + kk] = (u16)f2bf(vv[j]);
    }
    __syncthreads();

    Frag bfr[2];
#pragma unroll
    for (int ci = 0; ci < 2; ++ci) {
      int col = wn * 32 + ci * 16 + l15;
      bfr[ci].u[0] = lB[col * 4 + 2 * hi];
      bfr[ci].u[1] = lB[col * 4 + 2 * hi + 1];
    }
#pragma unroll
    for (int ri = 0; ri < 4; ++ri) {
      Frag a;
      int row = wm * 64 + ri * 16 + l15;
      a.u[0] = lA[row * 4 + 2 * hi];
      a.u[1] = lA[row * 4 + 2 * hi + 1];
#pragma unroll
      for (int ci = 0; ci < 2; ++ci) acc[ri][ci] = wmma_bf16(a, bfr[ci], acc[ri][ci]);
    }
  }

#pragma unroll
  for (int ri = 0; ri < 4; ++ri) {
#pragma unroll
    for (int ci = 0; ci < 2; ++ci) {
      int gn = n0 + wn * 32 + ci * 16 + l15;
      float bb = bias[gn];
#pragma unroll
      for (int r = 0; r < 8; ++r) {
        int gm = m0 + wm * 64 + ri * 16 + r + 8 * hi;
        float val = fmaxf(acc[ri][ci][r] + bb, 0.f);
        out[(size_t)gm * kD + gn] = val;
      }
    }
  }
}

// ---------------------------------------------------------------------------
extern "C" void kernel_launch(void* const* d_in, const int* in_sizes, int n_in,
                              void* d_out, int out_size, void* d_ws, size_t ws_size,
                              hipStream_t stream) {
  const float* q  = (const float*)d_in[0];
  const float* k  = (const float*)d_in[1];
  const float* v  = (const float*)d_in[2];
  // d_in[3] = mask: exact causal upper-triangle, handled analytically in-kernel
  const float* Wq = (const float*)d_in[4];
  const float* bq = (const float*)d_in[5];
  const float* Wk = (const float*)d_in[6];
  const float* bk = (const float*)d_in[7];
  const float* Wv = (const float*)d_in[8];
  const float* bv = (const float*)d_in[9];
  const float* Wo = (const float*)d_in[10];
  const float* bo = (const float*)d_in[11];
  float* out = (float*)d_out;

  // Workspace: Qh | Kh | Vh (bf16 [B,H,S,DK]) | attnW (bf16 [B,S,D])
  const size_t HE = (size_t)kB * kH * kS * kDK;  // 3,145,728 elems
  u16* Qh    = (u16*)d_ws;
  u16* Kh    = Qh + HE;
  u16* Vh    = Kh + HE;
  u16* attnW = Vh + HE;

  dim3 blk(256);
  dim3 gp(kD / 128, (kB * kS) / 128);  // (6, 32)
  proj_qkv_kernel<<<gp, blk, 0, stream>>>(q, Wq, bq, Qh);
  proj_qkv_kernel<<<gp, blk, 0, stream>>>(k, Wk, bk, Kh);
  proj_qkv_kernel<<<gp, blk, 0, stream>>>(v, Wv, bv, Vh);

  dim3 ga(kS / 128, kB * kH);  // (16, 24)
  attn_kernel<<<ga, blk, 0, stream>>>(Qh, Kh, Vh, attnW);

  out_proj_kernel<<<gp, blk, 0, stream>>>(attnW, Wo, bo, out);
}
